// GCNDecoder_32959579030036
// MI455X (gfx1250) — matrix operand order repaired and checked by reading, not compiled, not run on hardware
//
#include <hip/hip_runtime.h>
#include <hip/hip_bf16.h>

typedef __attribute__((ext_vector_type(16))) __bf16 v16bf;
typedef __attribute__((ext_vector_type(8)))  float  v8f;

#define GCN_N_NODES 100000
#define GCN_IN_C    64
#define GCN_HID_C   128
#define GCN_OUT_C   64

// ---------------------------------------------------------------------------
// Degree / normalization helpers
// ---------------------------------------------------------------------------
__global__ __launch_bounds__(256) void gcn_fill_ones(float* __restrict__ p, int n) {
    int i = blockIdx.x * blockDim.x + threadIdx.x;
    if (i < n) p[i] = 1.0f;  // self-loop contributes 1 to every node's degree
}

__global__ __launch_bounds__(256) void gcn_deg_accum(const int* __restrict__ dst,
                                                     float* __restrict__ deg, int E) {
    int i = blockIdx.x * blockDim.x + threadIdx.x;
    if (i < E) unsafeAtomicAdd(&deg[dst[i]], 1.0f);
}

__global__ __launch_bounds__(256) void gcn_rsqrt_ip(float* __restrict__ p, int n) {
    int i = blockIdx.x * blockDim.x + threadIdx.x;
    if (i < n) p[i] = rsqrtf(p[i]);  // deg >= 1 always (self-loops)
}

// ---------------------------------------------------------------------------
// Dense layer: D[M x NC] = op(A)[M x K] @ W[K x NC], bf16 WMMA, f32 accum.
// One wave computes one 16x16 output tile. M must be a multiple of 16.
// FUSE_RELU_BIAS: A element -> max(A + bias[k], 0) (GCN layer-2 input).
// ---------------------------------------------------------------------------
template <int K, int NC, bool FUSE_RELU_BIAS>
__global__ __launch_bounds__(256)
void gcn_gemm_bf16_wmma(const float* __restrict__ A,
                        const float* __restrict__ W,
                        const float* __restrict__ bias,
                        float* __restrict__ D,
                        int M) {
    const int lane    = threadIdx.x & 31;
    const int wave    = threadIdx.x >> 5;
    const int tilesN  = NC / 16;
    const int tile    = blockIdx.x * 8 + wave;
    const int numTile = (M / 16) * tilesN;
    if (tile >= numTile) return;  // grids are exact; whole waves only

    const int tm = tile / tilesN;
    const int tn = tile % tilesN;

    const int lo     = lane & 15;
    const int kShift = (lane < 16) ? 0 : 8;   // A-matrix K half (ISA 16-bit A layout)
    const int bShift = (lane < 16) ? 0 : 8;   // B-matrix K half (same as A)
    const int mrow   = tm * 16 + lo;          // A row this lane carries
    const int ncol   = tn * 16 + lo;          // B/D column this lane carries

    const float* __restrict__ arow = A + (size_t)mrow * K;

    v8f acc = {};
#pragma unroll
    for (int kb = 0; kb < K; kb += 32) {
        v16bf a, b;
#pragma unroll
        for (int j = 0; j < 8; ++j) {
            float v0 = arow[kb + kShift + j];           // VGPR0-3: K = kShift+0..7
            float v1 = arow[kb + 16 + kShift + j];      // VGPR4-7: K = 16+kShift+0..7
            if (FUSE_RELU_BIAS) {
                v0 = fmaxf(v0 + bias[kb + kShift + j], 0.0f);
                v1 = fmaxf(v1 + bias[kb + 16 + kShift + j], 0.0f);
            }
            a[j]     = (__bf16)v0;
            a[8 + j] = (__bf16)v1;
        }
#pragma unroll
        for (int j = 0; j < 16; ++j) {                  // column ncol, K = bShift+0..7, 16+bShift+0..7
            b[j] = (__bf16)W[(size_t)(kb + bShift + (j < 8 ? j : j + 8)) * NC + ncol];
        }
        acc = __builtin_amdgcn_wmma_f32_16x16x32_bf16(
            /*neg_a=*/false, a, /*neg_b=*/false, b,
            /*c_mod=*/(short)0, acc, /*reuse_a=*/false, /*reuse_b=*/false);
    }

    // C/D layout: lanes 0-15 rows 0..7, lanes 16-31 rows 8..15 (VGPR r -> row)
    const int mbase = tm * 16 + ((lane < 16) ? 0 : 8);
#pragma unroll
    for (int r = 0; r < 8; ++r)
        D[(size_t)(mbase + r) * NC + ncol] = acc[r];
}

// ---------------------------------------------------------------------------
// Self-loop + init pass: out[i,c] = feat[i,c]*dinv[i]^2 (+ bias[c]).
// Doubles as the zero/poison-clearing initializer of the accumulator.
// ---------------------------------------------------------------------------
template <int C, bool HAS_BIAS>
__global__ __launch_bounds__(256)
void gcn_init_selfloop(const float* __restrict__ feat,
                       const float* __restrict__ dinv,
                       const float* __restrict__ bias,
                       float* __restrict__ out, int n) {
    const int CH4 = C / 4;
    int gid = blockIdx.x * blockDim.x + threadIdx.x;
    if (gid >= n * CH4) return;
    int node = gid / CH4;
    int c4   = (gid % CH4) * 4;
    float di = dinv[node];
    float w  = di * di;
    const float4 v = *(const float4*)(feat + (size_t)node * C + c4);
    float4 o;
    o.x = v.x * w; o.y = v.y * w; o.z = v.z * w; o.w = v.w * w;
    if (HAS_BIAS) {
        const float4 bb = *(const float4*)(bias + c4);
        o.x += bb.x; o.y += bb.y; o.z += bb.z; o.w += bb.w;
    }
    *(float4*)(out + (size_t)node * C + c4) = o;
}

// ---------------------------------------------------------------------------
// Edge scatter: out[dst] += feat[src] * dinv[src]*dinv[dst].
// Thread = (edge, 4 channels); 32 consecutive threads cover one edge's 128 ch.
// 51 MB feature arrays are L2-resident (192 MB L2) -> atomics never hit HBM.
// ---------------------------------------------------------------------------
template <int C>
__global__ __launch_bounds__(256)
void gcn_scatter_edges(const int* __restrict__ src,
                       const int* __restrict__ dst,
                       const float* __restrict__ dinv,
                       const float* __restrict__ feat,
                       float* __restrict__ out, int E) {
    const int CH4 = C / 4;
    long long gid   = (long long)blockIdx.x * blockDim.x + threadIdx.x;
    long long total = (long long)E * CH4;
    if (gid >= total) return;
    int e  = (int)(gid / CH4);
    int c4 = (int)(gid % CH4) * 4;
    int s = src[e], d = dst[e];
    float nrm = dinv[s] * dinv[d];
    const float4 v = *(const float4*)(feat + (size_t)s * C + c4);
    float* o = out + (size_t)d * C + c4;
    unsafeAtomicAdd(o + 0, v.x * nrm);   // global_atomic_add_f32, no-return
    unsafeAtomicAdd(o + 1, v.y * nrm);
    unsafeAtomicAdd(o + 2, v.z * nrm);
    unsafeAtomicAdd(o + 3, v.w * nrm);
}

// ---------------------------------------------------------------------------
extern "C" void kernel_launch(void* const* d_in, const int* in_sizes, int n_in,
                              void* d_out, int out_size, void* d_ws, size_t ws_size,
                              hipStream_t stream) {
    const float* x   = (const float*)d_in[0];
    const int*   ei  = (const int*)d_in[1];   // [2, E] flat: src row then dst row
    const float* W3  = (const float*)d_in[2];
    const float* b3  = (const float*)d_in[3];
    const float* W4  = (const float*)d_in[4];
    const float* b4  = (const float*)d_in[5];
    float*       out = (float*)d_out;

    const int N = GCN_N_NODES;
    const int E = in_sizes[1] / 2;
    const int* src = ei;
    const int* dst = ei + E;

    // Workspace carve-up (256B aligned): dinv | t (t1 reused as t2) | agg1
    char* ws = (char*)d_ws;
    size_t off = 0;
    auto carve = [&](size_t bytes) -> void* {
        void* p = ws + off;
        off = (off + bytes + 255) & ~(size_t)255;
        return p;
    };
    float* dinv = (float*)carve((size_t)N * sizeof(float));
    float* t    = (float*)carve((size_t)N * GCN_HID_C * sizeof(float));
    float* agg1 = (float*)carve((size_t)N * GCN_HID_C * sizeof(float));
    (void)ws_size;

    const int B = 256;
    // 1) degrees (self-loop=1) -> dinv
    gcn_fill_ones<<<(N + B - 1) / B, B, 0, stream>>>(dinv, N);
    gcn_deg_accum<<<(E + B - 1) / B, B, 0, stream>>>(dst, dinv, E);
    gcn_rsqrt_ip<<<(N + B - 1) / B, B, 0, stream>>>(dinv, N);

    // 2) t1 = x @ W3   (100000x64 @ 64x128) -- WMMA bf16
    {
        int tiles = (N / 16) * (GCN_HID_C / 16);           // 50000 waves
        gcn_gemm_bf16_wmma<GCN_IN_C, GCN_HID_C, false>
            <<<tiles / 8, B, 0, stream>>>(x, W3, nullptr, t, N);
    }
    // 3) agg1 = selfloop-init, then scatter edges
    gcn_init_selfloop<GCN_HID_C, false>
        <<<(N * (GCN_HID_C / 4) + B - 1) / B, B, 0, stream>>>(t, dinv, nullptr, agg1, N);
    {
        long long total = (long long)E * (GCN_HID_C / 4);
        gcn_scatter_edges<GCN_HID_C>
            <<<(int)((total + B - 1) / B), B, 0, stream>>>(src, dst, dinv, t, agg1, E);
    }

    // 4) t2 = relu(agg1 + b3) @ W4  (100000x128 @ 128x64), bias+ReLU fused on A-load
    {
        int tiles = (N / 16) * (GCN_OUT_C / 16);           // 25000 waves
        gcn_gemm_bf16_wmma<GCN_HID_C, GCN_OUT_C, true>
            <<<tiles / 8, B, 0, stream>>>(agg1, W4, b3, t, N);
    }
    // 5) out = b4 + selfloop, then scatter edges
    gcn_init_selfloop<GCN_OUT_C, true>
        <<<(N * (GCN_OUT_C / 4) + B - 1) / B, B, 0, stream>>>(t, dinv, b4, out, N);
    {
        long long total = (long long)E * (GCN_OUT_C / 4);
        gcn_scatter_edges<GCN_OUT_C>
            <<<(int)((total + B - 1) / B), B, 0, stream>>>(src, dst, dinv, t, out, E);
    }
    (void)n_in; (void)out_size;
}
